// VQVAE_79551384257109
// MI455X (gfx1250) — matrix operand order repaired and checked by reading, not compiled, not run on hardware
//
#include <hip/hip_runtime.h>
#include <hip/hip_bf16.h>

typedef __attribute__((ext_vector_type(16))) __bf16 bf16x16;
typedef __attribute__((ext_vector_type(8)))  float  v8f;
typedef __attribute__((ext_vector_type(4)))  unsigned int u32x4;
typedef __attribute__((ext_vector_type(8)))  int   i32x8;
typedef __attribute__((ext_vector_type(4)))  int   i32x4;

#define KCODES 512
#define DIM    64
#define NROWS  (32 * 64 * 64)     // 131072
#define COLT   (KCODES / 16)      // 32 column tiles
#define NTILES (NROWS / 16)       // 8192 row tiles

#define PACK_BF16   (COLT * 2 * 32 * 16)            // 32768 bf16 per of hi/lo
#define STAGE_BYTES (PACK_BF16 * 2 * 2 + KCODES * 4) // 133120 bytes, qword-divisible

// ---------------------------------------------------------------------------
// Prep: pack codebook into WMMA B-fragment layout (split bf16 hi/lo),
// compute |e_k|^2, zero the loss accumulator.
// B fragment for (colTile t, k-chunk c): lane l holds column N = l&15,
// K = c*32 + (l<16 ? 0..15 : 16..31), 16 contiguous bf16 per lane.
// ---------------------------------------------------------------------------
__global__ __launch_bounds__(256) void vq_prep(const float* __restrict__ cb,
                                               __bf16* __restrict__ Bhi,
                                               __bf16* __restrict__ Blo,
                                               float* __restrict__ eknorm,
                                               float* __restrict__ lossAcc) {
    int tid = blockIdx.x * blockDim.x + threadIdx.x;
    if (tid < COLT * 2 * 32) {          // 2048 pack tasks
        int l = tid & 31;
        int c = (tid >> 5) & 1;
        int t = tid >> 6;
        int n  = t * 16 + (l & 15);
        int kb = c * 32 + ((l < 16) ? 0 : 16);
        const float* src = cb + n * DIM + kb;
        #pragma unroll
        for (int i = 0; i < 16; ++i) {
            float x = src[i];
            __bf16 h = (__bf16)x;
            float  r = x - (float)h;
            Bhi[tid * 16 + i] = h;
            Blo[tid * 16 + i] = (__bf16)r;
        }
    }
    if (tid < KCODES) {
        const float* src = cb + tid * DIM;
        float s = 0.f;
        #pragma unroll
        for (int i = 0; i < DIM; ++i) s += src[i] * src[i];
        eknorm[tid] = s;
    }
    if (tid == 0) *lossAcc = 0.f;
}

// Shared staging mirrors the ws pack layout exactly (one contiguous TDM copy).
struct __align__(128) Staging {
    __bf16 bhi[PACK_BF16];
    __bf16 blo[PACK_BF16];
    float  eknorm[KCODES];
};

// ---------------------------------------------------------------------------
// Main: one wave32 per 16-row tile; 8 waves/block. Codebook fragments are
// DMA'd into LDS once per block by the Tensor Data Mover, then the hot loop
// runs split-precision bf16 WMMA + register argmin entirely out of LDS.
// ---------------------------------------------------------------------------
__global__ __launch_bounds__(256) void vq_main(const float* __restrict__ z,
                                               const float* __restrict__ cb,
                                               const __bf16* __restrict__ packBase,
                                               float* __restrict__ lossAcc,
                                               float* __restrict__ out) {
    __shared__ Staging sm;

    const int lane = threadIdx.x & 31;
    const int wave = threadIdx.x >> 5;
    const int tile = blockIdx.x * 8 + wave;
    const int row0 = tile * 16;
    const int m    = lane & 15;     // A: row-in-tile; D: column-in-tile
    const int hi   = lane >> 4;

    // ---- TDM: one DMA of the packed codebook region (133120 B) into LDS ----
    if (wave == 0) {
        unsigned long long ga = (unsigned long long)(uintptr_t)packBase;
        unsigned int ldsOff   = (unsigned int)(uintptr_t)&sm;  // low 32b = LDS offset
        const unsigned int nq = STAGE_BYTES / 8u;              // 16640 qwords

        u32x4 g0;
        g0[0] = 1u;                                            // count=1, user mode
        g0[1] = ldsOff;                                        // lds_addr
        g0[2] = (unsigned int)(ga & 0xffffffffu);              // global_addr[31:0]
        g0[3] = (unsigned int)((ga >> 32) & 0x01ffffffu)       // global_addr[56:32]
              | (2u << 30);                                    // type=2 (image)

        i32x8 g1;
        g1[0] = (int)(3u << 16);                               // data_size=8B
        g1[1] = (int)((nq & 0xffffu) << 16);                   // tensor_dim0[15:0]
        g1[2] = (int)(((nq >> 16) & 0xffffu) | (1u << 16));    // td0[31:16] | tensor_dim1=1
        g1[3] = (int)((nq & 0xffffu) << 16);                   // tile_dim0 = nq
        g1[4] = 1;                                             // tile_dim1=1, tile_dim2=0
        g1[5] = (int)nq;                                       // tensor_dim0_stride lo
        g1[6] = (int)((nq & 0xffffu) << 16);                   // stride0 hi=0 | stride1 lo
        g1[7] = (int)((nq >> 16) & 0xffffu);                   // stride1 hi

        i32x4 gz = {0, 0, 0, 0};
#if __clang_major__ >= 23
        i32x8 gz8 = {0, 0, 0, 0, 0, 0, 0, 0};
        __builtin_amdgcn_tensor_load_to_lds(g0, g1, gz, gz, gz8, 0);
#else
        __builtin_amdgcn_tensor_load_to_lds(g0, g1, gz, gz, 0);
#endif
        __builtin_amdgcn_s_wait_tensorcnt(0);
    }
    __syncthreads();

    // ---- Build A fragments (16x32 bf16, contiguous k order, same as B pack), split hi/lo ----
    const float* zrow = z + (size_t)(row0 + m) * DIM;
    bf16x16 ah[2], al[2];
    #pragma unroll
    for (int c = 0; c < 2; ++c) {
        int kb = c * 32 + hi * 16;  // lane<16: K{0..15}; lane>=16: K{16..31}
        float tmp[16];
        #pragma unroll
        for (int i = 0; i < 8; ++i) tmp[i]     = zrow[kb + i];
        #pragma unroll
        for (int i = 0; i < 8; ++i) tmp[8 + i] = zrow[kb + 8 + i];
        #pragma unroll
        for (int i = 0; i < 16; ++i) {
            __bf16 h = (__bf16)tmp[i];
            ah[c][i] = h;
            al[c][i] = (__bf16)(tmp[i] - (float)h);
        }
    }

    float bestV[8];
    int   bestI[8];
    #pragma unroll
    for (int j = 0; j < 8; ++j) { bestV[j] = 3.4e38f; bestI[j] = 0; }

    // ---- Sweep 32 column tiles: 6 bf16 WMMAs each, operands from LDS ----
    for (int t = 0; t < COLT; ++t) {
        const bf16x16 bh0 = *(const bf16x16*)(sm.bhi + ((t * 2 + 0) * 32 + lane) * 16);
        const bf16x16 bl0 = *(const bf16x16*)(sm.blo + ((t * 2 + 0) * 32 + lane) * 16);
        const bf16x16 bh1 = *(const bf16x16*)(sm.bhi + ((t * 2 + 1) * 32 + lane) * 16);
        const bf16x16 bl1 = *(const bf16x16*)(sm.blo + ((t * 2 + 1) * 32 + lane) * 16);
        v8f acc = {};
        acc = __builtin_amdgcn_wmma_f32_16x16x32_bf16(false, ah[0], false, bh0, (short)0, acc, false, false);
        acc = __builtin_amdgcn_wmma_f32_16x16x32_bf16(false, ah[0], false, bl0, (short)0, acc, false, false);
        acc = __builtin_amdgcn_wmma_f32_16x16x32_bf16(false, al[0], false, bh0, (short)0, acc, false, false);
        acc = __builtin_amdgcn_wmma_f32_16x16x32_bf16(false, ah[1], false, bh1, (short)0, acc, false, false);
        acc = __builtin_amdgcn_wmma_f32_16x16x32_bf16(false, ah[1], false, bl1, (short)0, acc, false, false);
        acc = __builtin_amdgcn_wmma_f32_16x16x32_bf16(false, al[1], false, bh1, (short)0, acc, false, false);

        // d(row, col) = |e|^2 - 2*dot  (|z|^2 constant per row -> irrelevant)
        float en = sm.eknorm[t * 16 + m];       // column N = m
        #pragma unroll
        for (int j = 0; j < 8; ++j) {
            float dist = en - 2.0f * acc[j];    // D VGPR j: row = j + 8*hi
            if (dist < bestV[j]) { bestV[j] = dist; bestI[j] = t * 16 + m; }
        }
    }

    // ---- Argmin across the 16 columns held in each 16-lane half ----
    #pragma unroll
    for (int j = 0; j < 8; ++j) {
        float v = bestV[j]; int idx = bestI[j];
        #pragma unroll
        for (int off = 1; off < 16; off <<= 1) {
            float ov = __shfl_xor(v, off, 32);
            int   oi = __shfl_xor(idx, off, 32);
            if (ov < v || (ov == v && oi < idx)) { v = ov; idx = oi; }
        }
        bestI[j] = idx;     // uniform across the 16-lane group
    }

    // ---- Gather winning codes (exact f32), write output, loss partial ----
    float lsum = 0.f;
    #pragma unroll
    for (int j = 0; j < 8; ++j) {
        int r = row0 + j + 8 * hi;
        const float* e  = cb + (size_t)bestI[j] * DIM;
        const float* zr = z  + (size_t)r * DIM;
        float4 ev = *(const float4*)(e  + m * 4);
        float4 zv = *(const float4*)(zr + m * 4);
        *(float4*)(out + (size_t)r * DIM + m * 4) = ev;
        float dx = ev.x - zv.x, dy = ev.y - zv.y;
        float dz2 = ev.z - zv.z, dw = ev.w - zv.w;
        lsum += dx * dx + dy * dy + dz2 * dz2 + dw * dw;
    }
    #pragma unroll
    for (int off = 1; off < 32; off <<= 1) lsum += __shfl_xor(lsum, off, 32);
    if (lane == 0) atomicAdd(lossAcc, lsum);
}

__global__ void vq_finish(const float* __restrict__ lossAcc,
                          float* __restrict__ out, int outN) {
    // loss = q_latent + 0.25*e_latent = 1.25 * mean((q - z)^2)
    out[outN] = 1.25f * (*lossAcc) / (float)(NROWS * DIM);
}

// ---------------------------------------------------------------------------
extern "C" void kernel_launch(void* const* d_in, const int* in_sizes, int n_in,
                              void* d_out, int out_size, void* d_ws, size_t ws_size,
                              hipStream_t stream) {
    const float* z  = (const float*)d_in[0];   // [32,64,64,64] f32
    const float* cb = (const float*)d_in[1];   // [512,64] f32
    float* out = (float*)d_out;                // [131072*64] quantized + [1] loss

    // Workspace layout (contiguous, one TDM copy): Bhi | Blo | eknorm | lossAcc
    __bf16* Bhi     = (__bf16*)d_ws;
    __bf16* Blo     = Bhi + PACK_BF16;
    float*  eknorm  = (float*)(Blo + PACK_BF16);
    float*  lossAcc = eknorm + KCODES;

    vq_prep<<<8, 256, 0, stream>>>(cb, Bhi, Blo, eknorm, lossAcc);
    vq_main<<<NTILES / 8, 256, 0, stream>>>(z, cb, Bhi, lossAcc, out);
    vq_finish<<<1, 1, 0, stream>>>(lossAcc, out, out_size - 1);
}
